// MultiHeadAttentionLayer_678604832906
// MI455X (gfx1250) — compile-verified, matrix operand order checked by reading, not run on hardware
//
#include <hip/hip_runtime.h>
#include <hip/hip_bf16.h>

typedef __attribute__((ext_vector_type(16))) __bf16 v16bf;
typedef __attribute__((ext_vector_type(8)))  float  v8f;

#define IN_DIM     128
#define HEADS      8
#define D_HEAD     16
#define OUT_DIM    128           // HEADS*D_HEAD
#define QKV_STRIDE 384           // Q|K|V concatenated per node
#define PART_ELEMS (4*8*32*16)   // frag elems per weight matrix: ks*ct*lane*16

// ---------------------------------------------------------------------------
// A-fragment loader: 16x32 bf16 tile of a row-major f32 matrix.
// Lane l holds row m=l&15; k-base kb = ks*32 + (l>>4)*8; elements 0..7 cover
// k=kb..kb+7, elements 8..15 cover k=kb+16..kb+23 (per ISA 16-bit A layout).
// ---------------------------------------------------------------------------
__device__ __forceinline__ v16bf load_a_frag(const float* __restrict__ rowp, int kb) {
  const float4* rp = (const float4*)(rowp + kb);
  float4 f0 = rp[0], f1 = rp[1], f2 = rp[4], f3 = rp[5];
  v16bf a;
  a[0]=(__bf16)f0.x;  a[1]=(__bf16)f0.y;  a[2]=(__bf16)f0.z;  a[3]=(__bf16)f0.w;
  a[4]=(__bf16)f1.x;  a[5]=(__bf16)f1.y;  a[6]=(__bf16)f1.z;  a[7]=(__bf16)f1.w;
  a[8]=(__bf16)f2.x;  a[9]=(__bf16)f2.y;  a[10]=(__bf16)f2.z; a[11]=(__bf16)f2.w;
  a[12]=(__bf16)f3.x; a[13]=(__bf16)f3.y; a[14]=(__bf16)f3.z; a[15]=(__bf16)f3.w;
  return a;
}

// ---------------------------------------------------------------------------
// Kernel 0: pack weights [128,128] f32 (k-major) -> bf16 B-fragments stored in
// fragment order: frag[(((part*4+ks)*8+ct)*32+lane)*16 + i]. parts: Wq,Wk,Wv,We
// B-frag (32x16): lane l -> col n=l&15, kb=(l>>4)*8; elem i -> k=kb+i(+8 if i>=8)
// ---------------------------------------------------------------------------
__global__ void convert_weights_kernel(const float* __restrict__ Wq,
                                       const float* __restrict__ Wk,
                                       const float* __restrict__ Wv,
                                       const float* __restrict__ We,
                                       __bf16* __restrict__ frag) {
  int gid  = blockIdx.x * blockDim.x + threadIdx.x;   // 0..4095
  int lane = gid & 31;
  int ct   = (gid >> 5) & 7;
  int ks   = (gid >> 8) & 3;
  int part = gid >> 10;
  const float* W = (part == 0) ? Wq : (part == 1) ? Wk : (part == 2) ? Wv : We;
  int n  = ct * 16 + (lane & 15);
  int kb = ks * 32 + ((lane >> 4) << 3);
  __bf16* dst = frag + (size_t)gid * 16;
  #pragma unroll
  for (int i = 0; i < 16; ++i) {
    int k = kb + i + ((i >= 8) ? 8 : 0);
    dst[i] = (__bf16)W[k * OUT_DIM + n];
  }
}

// ---------------------------------------------------------------------------
// Zero the scatter targets (d_out used as wV accumulator, Z denominators).
// ---------------------------------------------------------------------------
__global__ void zero_kernel(float* __restrict__ out, float* __restrict__ Z,
                            int n_out, int n_z) {
  int i = blockIdx.x * blockDim.x + threadIdx.x;
  if (i < n_out) out[i] = 0.0f;
  if (i < n_z)   Z[i]   = 0.0f;
}

// ---------------------------------------------------------------------------
// Kernel 1: node projections QKV = x @ [Wq|Wk|Wv] + bias  -> [N, 384] f32
// One wave per 16-row tile; 3 parts x 8 ctiles x 4 ksteps = 96 WMMAs.
// Weight fragments staged in LDS (96 KB) once per workgroup; B-frags consumed
// in pairs to bound register pressure (no spills).
// ---------------------------------------------------------------------------
__global__ void node_proj_kernel(const float* __restrict__ x,
                                 const __bf16* __restrict__ frag,
                                 const float* __restrict__ bq,
                                 const float* __restrict__ bk,
                                 const float* __restrict__ bv,
                                 float* __restrict__ qkv,
                                 int numTiles) {
  __shared__ __bf16 sW[3 * PART_ELEMS];          // 96 KB
  {
    const uint4* srcp = (const uint4*)frag;
    uint4* dstp = (uint4*)sW;
    #pragma unroll
    for (int i = 0; i < 12; ++i)                 // 3*PART_ELEMS*2B / 16B = 6144
      dstp[threadIdx.x + i * 256] = srcp[threadIdx.x + i * 256];
  }
  __syncthreads();

  int wave = (int)((blockIdx.x * blockDim.x + threadIdx.x) >> 5);
  if (wave >= numTiles) return;
  int lane = threadIdx.x & 31;
  int hi   = lane >> 4;
  int d    = lane & 15;
  int row0 = wave * 16;

  const float* rowp = x + (size_t)(row0 + (lane & 15)) * IN_DIM;

  #pragma unroll
  for (int part = 0; part < 3; ++part) {
    const float* bias = (part == 0) ? bq : (part == 1) ? bk : bv;
    v8f acc[8];
    #pragma unroll
    for (int ct = 0; ct < 8; ++ct) {
      float b0 = bias[ct * 16 + d];
      #pragma unroll
      for (int r = 0; r < 8; ++r) acc[ct][r] = b0;
    }
    #pragma unroll
    for (int ks = 0; ks < 4; ++ks) {
      v16bf a = load_a_frag(rowp, ks * 32 + (hi << 3));
      const __bf16* bbase = sW + (((size_t)part * 4 + ks) * 8) * 32 * 16
                               + (size_t)lane * 16;
      #pragma unroll
      for (int ct = 0; ct < 8; ct += 2) {
        v16bf b0 = *(const v16bf*)(bbase + (size_t)(ct + 0) * 32 * 16);
        v16bf b1 = *(const v16bf*)(bbase + (size_t)(ct + 1) * 32 * 16);
        acc[ct + 0] = __builtin_amdgcn_wmma_f32_16x16x32_bf16(
            false, a, false, b0, (short)0, acc[ct + 0], false, false);
        acc[ct + 1] = __builtin_amdgcn_wmma_f32_16x16x32_bf16(
            false, a, false, b1, (short)0, acc[ct + 1], false, false);
      }
    }
    #pragma unroll
    for (int ct = 0; ct < 8; ++ct) {
      #pragma unroll
      for (int r = 0; r < 8; ++r) {
        qkv[(size_t)(row0 + r + 8 * hi) * QKV_STRIDE + part * OUT_DIM + ct * 16 + d]
            = acc[ct][r];
      }
    }
  }
}

// ---------------------------------------------------------------------------
// Kernel 2 (fused): Eh = edge_attr @ We + be kept in WMMA accumulators;
// per-edge per-head score via 16-lane butterfly reduce; atomic scatter of
// weighted V and score. C-frag layout makes ctile==head, lane n==head dim.
// We fragments staged in LDS (32 KB); B-frags consumed in pairs.
// ---------------------------------------------------------------------------
__global__ void edge_attn_kernel(const float* __restrict__ edge_attr,
                                 const long long* __restrict__ eidx,
                                 const __bf16* __restrict__ fragWe,
                                 const float* __restrict__ be,
                                 const float* __restrict__ qkv,
                                 float* __restrict__ wV,   // d_out [N,128]
                                 float* __restrict__ Z,    // [N,8]
                                 long long E, int numTiles) {
  __shared__ __bf16 sWe[PART_ELEMS];             // 32 KB
  {
    const uint4* srcp = (const uint4*)fragWe;
    uint4* dstp = (uint4*)sWe;
    #pragma unroll
    for (int i = 0; i < 4; ++i)                  // PART_ELEMS*2B / 16B = 2048
      dstp[threadIdx.x + i * 256] = srcp[threadIdx.x + i * 256];
  }
  __syncthreads();

  int wave = (int)((blockIdx.x * blockDim.x + threadIdx.x) >> 5);
  if (wave >= numTiles) return;
  int lane = threadIdx.x & 31;
  int hi   = lane >> 4;
  int d    = lane & 15;
  long long e0 = (long long)wave * 16;

  // --- Eh GEMM: 16 edges x 128 cols, bf16 WMMA, f32 accumulate -------------
  v8f acc[8];
  #pragma unroll
  for (int h = 0; h < 8; ++h) {
    float b0 = be[h * 16 + d];
    #pragma unroll
    for (int r = 0; r < 8; ++r) acc[h][r] = b0;
  }
  const float* rowp = edge_attr + (size_t)(e0 + (lane & 15)) * IN_DIM;
  #pragma unroll
  for (int ks = 0; ks < 4; ++ks) {
    v16bf a = load_a_frag(rowp, ks * 32 + (hi << 3));
    const __bf16* bbase = sWe + ((size_t)ks * 8) * 32 * 16 + (size_t)lane * 16;
    #pragma unroll
    for (int h = 0; h < 8; h += 2) {
      v16bf b0 = *(const v16bf*)(bbase + (size_t)(h + 0) * 32 * 16);
      v16bf b1 = *(const v16bf*)(bbase + (size_t)(h + 1) * 32 * 16);
      acc[h + 0] = __builtin_amdgcn_wmma_f32_16x16x32_bf16(
          false, a, false, b0, (short)0, acc[h + 0], false, false);
      acc[h + 1] = __builtin_amdgcn_wmma_f32_16x16x32_bf16(
          false, a, false, b1, (short)0, acc[h + 1], false, false);
    }
  }

  // --- score + scatter -----------------------------------------------------
  const float inv_sqrt_d = 0.25f;   // 1/sqrt(16)
  #pragma unroll
  for (int r = 0; r < 8; ++r) {
    long long e = e0 + r + 8 * hi;          // edge owned by this half-wave row
    int s = (int)eidx[e];
    int t = (int)eidx[E + e];
    const float* Qp = qkv + (size_t)t * QKV_STRIDE;
    const float* Kp = qkv + (size_t)s * QKV_STRIDE + OUT_DIM;
    const float* Vp = qkv + (size_t)s * QKV_STRIDE + 2 * OUT_DIM;
    float* Wp = wV + (size_t)t * OUT_DIM;
    #pragma unroll
    for (int h = 0; h < 8; ++h) {
      int c = h * 16 + d;
      float term = Kp[c] * Qp[c] * inv_sqrt_d * acc[h][r];
      term += __shfl_xor(term, 1);
      term += __shfl_xor(term, 2);
      term += __shfl_xor(term, 4);
      term += __shfl_xor(term, 8);          // butterfly within 16-lane half
      float sc = __expf(fminf(fmaxf(term, -5.0f), 5.0f));
      atomicAdd(&Wp[c], Vp[c] * sc);
      if (d == 0) atomicAdd(&Z[(size_t)t * HEADS + h], sc);
    }
  }
}

// ---------------------------------------------------------------------------
// Kernel 3: out = wV / (Z + 1e-6), in place in d_out.
// ---------------------------------------------------------------------------
__global__ void normalize_kernel(float* __restrict__ out,
                                 const float* __restrict__ Z, int n) {
  int i = blockIdx.x * blockDim.x + threadIdx.x;
  if (i >= n) return;
  int node = i >> 7;
  int h    = (i >> 4) & 7;
  out[i] = out[i] / (Z[(size_t)node * HEADS + h] + 1e-6f);
}

// ---------------------------------------------------------------------------
extern "C" void kernel_launch(void* const* d_in, const int* in_sizes, int n_in,
                              void* d_out, int out_size, void* d_ws, size_t ws_size,
                              hipStream_t stream) {
  const float*     x         = (const float*)d_in[0];
  const float*     edge_attr = (const float*)d_in[1];
  const long long* eidx      = (const long long*)d_in[2];
  const float*     Wq = (const float*)d_in[3];
  const float*     bq = (const float*)d_in[4];
  const float*     Wk = (const float*)d_in[5];
  const float*     bk = (const float*)d_in[6];
  const float*     We = (const float*)d_in[7];
  const float*     be = (const float*)d_in[8];
  const float*     Wv = (const float*)d_in[9];
  const float*     bv = (const float*)d_in[10];
  float* out = (float*)d_out;

  const int       N = in_sizes[0] / IN_DIM;
  const long long E = (long long)in_sizes[2] / 2;

  // workspace: qkv [N,384] f32 | Z [N,8] f32 | weight fragments (4 parts) bf16
  char* ws = (char*)d_ws;
  float* qkv = (float*)ws;
  size_t off = (size_t)N * QKV_STRIDE * sizeof(float);
  off = (off + 255) & ~(size_t)255;
  float* Z = (float*)(ws + off);
  off += (size_t)N * HEADS * sizeof(float);
  off = (off + 255) & ~(size_t)255;
  __bf16* frag = (__bf16*)(ws + off);

  convert_weights_kernel<<<16, 256, 0, stream>>>(Wq, Wk, Wv, We, frag);

  const int n_out = N * OUT_DIM;
  const int n_z   = N * HEADS;
  zero_kernel<<<(n_out + 255) / 256, 256, 0, stream>>>(out, Z, n_out, n_z);

  const int nodeTiles = (N + 15) / 16;
  node_proj_kernel<<<(nodeTiles + 7) / 8, 256, 0, stream>>>(
      x, frag, bq, bk, bv, qkv, nodeTiles);

  const int edgeTiles = (int)((E + 15) / 16);
  edge_attn_kernel<<<(edgeTiles + 7) / 8, 256, 0, stream>>>(
      edge_attr, eidx, frag + (size_t)3 * PART_ELEMS, be, qkv, out, Z, E, edgeTiles);

  normalize_kernel<<<(n_out + 255) / 256, 256, 0, stream>>>(out, Z, n_out);
}
